// MambaLayer_8985071583555
// MI455X (gfx1250) — matrix operand order repaired and checked by reading, not compiled, not run on hardware
//
#include <hip/hip_runtime.h>

// ---------------------------------------------------------------------------
// Types for CDNA5 WMMA (wave32): bf16 A/B fragments (16 bf16 = 8 VGPRs),
// f32 accumulator (8 VGPRs).
// ---------------------------------------------------------------------------
typedef __bf16 bf16_t;
typedef __attribute__((ext_vector_type(16))) __bf16      v16bf;
typedef __attribute__((ext_vector_type(8)))  float       v8f;
typedef __attribute__((ext_vector_type(4)))  unsigned int uint4v;
typedef __attribute__((ext_vector_type(4)))  int          int4v;

union FragU {            // 32-byte pun: two 16B chunks <-> one bf16x16 fragment
    uint4v q[2];
    v16bf  bf;
};

// Async global->LDS path (ASYNCcnt-tracked) if this toolchain exposes it.
// Probe-discovered signature: (v4i addrspace(1)*, v4i addrspace(3)*, imm, imm)
#if defined(__has_builtin)
#  if __has_builtin(__builtin_amdgcn_global_load_async_to_lds_b128) && \
      __has_builtin(__builtin_amdgcn_s_wait_asynccnt)
#    define USE_ASYNC_LDS 1
#  endif
#endif
#ifndef USE_ASYNC_LDS
#  define USE_ASYNC_LDS 0
#endif

__device__ __forceinline__ void stage_b128(const unsigned short* g,
                                           unsigned short* l) {
#if USE_ASYNC_LDS
    __builtin_amdgcn_global_load_async_to_lds_b128(
        (__attribute__((address_space(1))) int4v*)g,
        (__attribute__((address_space(3))) int4v*)l,
        0, 0);
#else
    *(uint4v*)l = *(const uint4v*)g;
#endif
}

__device__ __forceinline__ unsigned short f2bf(float f) {
    unsigned int u = __builtin_bit_cast(unsigned int, f);
    u += 0x7fffu + ((u >> 16) & 1u);     // round-to-nearest-even
    return (unsigned short)(u >> 16);
}

// A fragment: 16x32 (MxK) bf16. lane = 16*half + m.
// elements 0..15 -> K = kbase + 16*half + i         (32B contiguous)
__device__ __forceinline__ v16bf load_a_frag(const unsigned short* __restrict__ A,
                                             int lda, int mbase, int kbase, int lane) {
    int m    = mbase + (lane & 15);
    int half = lane >> 4;
    const unsigned short* p = A + (size_t)m * lda + kbase + 16 * half;
    FragU u;
    u.q[0] = *(const uint4v*)(p);
    u.q[1] = *(const uint4v*)(p + 8);
    return u.bf;
}

// B fragment: 32x16 (KxN) bf16 with B[k][n] = W[n][k]. lane = 16*half + n.
// elements 0..15 -> K = kbase + 16*half + i  => 32B contiguous along weight row.
__device__ __forceinline__ v16bf load_b_frag(const unsigned short* __restrict__ W,
                                             int ldb, int nbase, int kbase, int lane) {
    int n    = nbase + (lane & 15);
    int half = lane >> 4;
    const unsigned short* p = W + (size_t)n * ldb + kbase + 16 * half;
    FragU u;
    u.q[0] = *(const uint4v*)(p);
    u.q[1] = *(const uint4v*)(p + 8);
    return u.bf;
}

// ---------------------------------------------------------------------------
// LDS-staged GEMM: C[m][n] = sum_k A[m][k] * W[n][k].
// Block tile 128M x 64N (8 waves x 16M). The 64x32 bf16 B tile (4 KB) is
// staged into LDS (async double-buffered, shared by all 8 waves); each wave
// prefetches its next A fragment into registers while WMMAs run.
// Requires M%128==0, N%64==0, K%32==0.
// ---------------------------------------------------------------------------
__global__ void gemm_bf16_lds(const unsigned short* __restrict__ A,
                              const unsigned short* __restrict__ W,
                              float* __restrict__ C,
                              int M, int N, int K) {
    __shared__ unsigned short lB[2][64 * 32];

    const int nb   = N / 64;
    const int mt   = blockIdx.x / nb;
    const int nt   = blockIdx.x % nb;
    const int lane = threadIdx.x & 31;
    const int wave = threadIdx.x >> 5;

    // Staging map: 256 threads x 16B cover the 4 KB tile.
    const int stRow   = threadIdx.x >> 2;   // 0..63 (n within tile)
    const int stChunk = threadIdx.x & 3;    // 0..3  (k chunk of 8)
    const unsigned short* wsrc =
        W + (size_t)(nt * 64 + stRow) * K + stChunk * 8;
    const int stOff = stRow * 32 + stChunk * 8;

    const int KS   = K / 32;
    const int mrow = mt * 128 + wave * 16;

    v8f acc[4];
#pragma unroll
    for (int i = 0; i < 4; ++i)
#pragma unroll
        for (int e = 0; e < 8; ++e) acc[i][e] = 0.f;

    // Prologue: stage tile 0, prefetch A fragment 0.
    stage_b128(wsrc, &lB[0][stOff]);
    v16bf a_cur = load_a_frag(A, K, mrow, 0, lane);

    for (int ks = 0; ks < KS; ++ks) {
        const int  cur  = ks & 1;
        const bool more = (ks + 1) < KS;
        if (more)
            stage_b128(wsrc + (size_t)(ks + 1) * 32, &lB[cur ^ 1][stOff]);
#if USE_ASYNC_LDS
        if (more) __builtin_amdgcn_s_wait_asynccnt(1);   // tile ks landed
        else      __builtin_amdgcn_s_wait_asynccnt(0);
#endif
        __syncthreads();                                  // whole tile visible

        v16bf a = a_cur;
        if (more) a_cur = load_a_frag(A, K, mrow, (ks + 1) * 32, lane);

#pragma unroll
        for (int i = 0; i < 4; ++i) {
            const unsigned short* p =
                &lB[cur][(i * 16 + (lane & 15)) * 32 + 16 * (lane >> 4)];
            FragU u;
            u.q[0] = *(const uint4v*)(p);
            u.q[1] = *(const uint4v*)(p + 8);
            acc[i] = __builtin_amdgcn_wmma_f32_16x16x32_bf16(
                false, a, false, u.bf, (short)0, acc[i], false, false);
        }
        __syncthreads();        // all reads of buf[cur] done before restaging
    }

    // C/D layout: VGPR r -> M = r + 8*half, N = lane&15.
    const int half = lane >> 4, n0 = lane & 15;
#pragma unroll
    for (int i = 0; i < 4; ++i) {
        int ncol = nt * 64 + i * 16 + n0;
#pragma unroll
        for (int r = 0; r < 8; ++r)
            C[(size_t)(mrow + 8 * half + r) * N + ncol] = acc[i][r];
    }
}

// ---------------------------------------------------------------------------
// Direct-load GEMM (for N not divisible by 64; x_proj N=112).
// One wave computes a 16 x (16*NB) strip.
// ---------------------------------------------------------------------------
template <int NB>
__global__ void gemm_bf16_nt(const unsigned short* __restrict__ A,
                             const unsigned short* __restrict__ W,
                             float* __restrict__ C,
                             int M, int N, int K) {
    int lane = threadIdx.x & 31;
    int wave = threadIdx.x >> 5;
    int gw   = blockIdx.x * (blockDim.x >> 5) + wave;
    int ntg  = N / (16 * NB);
    int mt   = gw / ntg;
    int ng   = gw % ntg;
    if (mt >= M / 16) return;

    v8f acc[NB];
#pragma unroll
    for (int i = 0; i < NB; ++i)
#pragma unroll
        for (int e = 0; e < 8; ++e) acc[i][e] = 0.f;

    for (int k = 0; k < K; k += 32) {
        v16bf a = load_a_frag(A, K, mt * 16, k, lane);
#pragma unroll
        for (int i = 0; i < NB; ++i) {
            v16bf b = load_b_frag(W, K, (ng * NB + i) * 16, k, lane);
            acc[i] = __builtin_amdgcn_wmma_f32_16x16x32_bf16(
                false, a, false, b, (short)0, acc[i], false, false);
        }
    }

    int half = lane >> 4, n0 = lane & 15;
#pragma unroll
    for (int i = 0; i < NB; ++i) {
        int ncol = (ng * NB + i) * 16 + n0;
#pragma unroll
        for (int r = 0; r < 8; ++r) {
            int m = mt * 16 + 8 * half + r;
            C[(size_t)m * N + ncol] = acc[i][r];
        }
    }
}

// ---------------------------------------------------------------------------
// LayerNorm (one wave per 256-wide row) -> bf16 output for WMMA consumption.
// ---------------------------------------------------------------------------
__global__ void layernorm_bf16(const float* __restrict__ x,
                               const float* __restrict__ w,
                               const float* __restrict__ b,
                               unsigned short* __restrict__ out,
                               int rows, int C) {
    int lane = threadIdx.x & 31;
    int row  = (blockIdx.x * blockDim.x + threadIdx.x) >> 5;
    if (row >= rows) return;
    const float* xr = x + (size_t)row * C;
    float s = 0.f, s2 = 0.f;
    for (int c = lane; c < C; c += 32) { float v = xr[c]; s += v; s2 += v * v; }
#pragma unroll
    for (int off = 16; off > 0; off >>= 1) {
        s  += __shfl_xor(s,  off, 32);
        s2 += __shfl_xor(s2, off, 32);
    }
    float mu   = s / C;
    float var  = s2 / C - mu * mu;
    float rstd = rsqrtf(var + 1e-5f);
    unsigned short* orow = out + (size_t)row * C;
    for (int c = lane; c < C; c += 32)
        orow[c] = f2bf((xr[c] - mu) * rstd * w[c] + b[c]);
}

__global__ void f32_to_bf16(const float* __restrict__ in,
                            unsigned short* __restrict__ out, int n) {
    for (int i = blockIdx.x * blockDim.x + threadIdx.x; i < n;
         i += gridDim.x * blockDim.x)
        out[i] = f2bf(in[i]);
}

// ---------------------------------------------------------------------------
// Causal depthwise conv1d (kernel 4) + SiLU. xz rows are 1024 wide (xi | z).
// Writes u in f32 (scan) and bf16 (x_proj GEMM).
// ---------------------------------------------------------------------------
__global__ void conv_silu(const float* __restrict__ xz,
                          const float* __restrict__ cw,
                          const float* __restrict__ cb,
                          float* __restrict__ u_f32,
                          unsigned short* __restrict__ u_bf,
                          int B, int N) {
    const int DI = 512;
    int idx = blockIdx.x * blockDim.x + threadIdx.x;
    if (idx >= B * N * DI) return;
    int d = idx % DI;
    int n = (idx / DI) % N;
    int b = idx / (DI * N);
    float acc = cb[d];
#pragma unroll
    for (int j = 0; j < 4; ++j) {
        int t = n - 3 + j;
        if (t >= 0)
            acc += xz[((size_t)b * N + t) * 1024 + d] * cw[d * 4 + j];
    }
    float s = acc / (1.f + __expf(-acc));   // SiLU
    u_f32[idx] = s;
    u_bf[idx]  = f2bf(s);
}

// ---------------------------------------------------------------------------
// dt = softplus(dtr @ dt_proj_w^T + b).  K=16: plain VALU dot product.
// ---------------------------------------------------------------------------
__global__ void dt_proj(const float* __restrict__ x_dbl,
                        const float* __restrict__ w,
                        const float* __restrict__ bias,
                        float* __restrict__ dt, int rows) {
    const int DI = 512, LD = 112;
    int idx = blockIdx.x * blockDim.x + threadIdx.x;
    if (idx >= rows * DI) return;
    int d = idx % DI;
    int m = idx / DI;
    const float* xr = x_dbl + (size_t)m * LD;
    float acc = bias[d];
#pragma unroll
    for (int r = 0; r < 16; ++r) acc += xr[r] * w[d * 16 + r];
    dt[idx] = (acc > 20.f) ? acc : log1pf(__expf(acc));
}

// ---------------------------------------------------------------------------
// Selective scan, serial over t. 16 lanes per channel d, 3 states per lane
// (48 = 16*3); h and A in registers; B_t/C_t broadcast via LDS; y reduced
// with 4 xor-shuffles; fused D-skip + SiLU(z) gate + bf16 cast for out_proj.
// Grid: B*32 blocks x 256 threads (16 channels per block).
// ---------------------------------------------------------------------------
__global__ void scan_kernel(const float* __restrict__ dt,
                            const float* __restrict__ x_dbl,
                            const float* __restrict__ u,
                            const float* __restrict__ A_log,
                            const float* __restrict__ Dvec,
                            const float* __restrict__ xz,
                            unsigned short* __restrict__ y_bf,
                            int B, int N) {
    const int DI = 512, DS = 48, LD = 112;
    int b     = blockIdx.x >> 5;            // 32 blocks per batch
    int dbase = (blockIdx.x & 31) * 16;
    int grp   = threadIdx.x >> 4;           // channel within block (0..15)
    int l     = threadIdx.x & 15;           // state-lane within channel
    int d     = dbase + grp;

    __shared__ float sB[48], sC[48];

    float h[3], Ad[3];
#pragma unroll
    for (int j = 0; j < 3; ++j) {
        int s = l + 16 * j;
        h[j]  = 0.f;
        Ad[j] = -__expf(A_log[(size_t)d * DS + s]);
    }
    float Dd = Dvec[d];

    for (int t = 0; t < N; ++t) {
        size_t row = (size_t)b * N + t;
        __syncthreads();                     // protect previous-step sB/sC reads
        if (threadIdx.x < 96) {
            int i = threadIdx.x;
            float v = x_dbl[row * LD + 16 + i];   // [16:64)=B, [64:112)=C
            if (i < 48) sB[i] = v; else sC[i - 48] = v;
        }
        __syncthreads();

        float dtv = dt[row * DI + d];
        float uv  = u[row * DI + d];
        float du  = dtv * uv;
        float y   = 0.f;
#pragma unroll
        for (int j = 0; j < 3; ++j) {
            int s = l + 16 * j;
            h[j] = __expf(dtv * Ad[j]) * h[j] + du * sB[s];
            y   += h[j] * sC[s];
        }
        y += __shfl_xor(y, 1, 32);
        y += __shfl_xor(y, 2, 32);
        y += __shfl_xor(y, 4, 32);
        y += __shfl_xor(y, 8, 32);
        if (l == 0) {
            float z  = xz[row * 1024 + 512 + d];
            float sz = z / (1.f + __expf(-z));
            y_bf[row * DI + d] = f2bf((y + uv * Dd) * sz);
        }
    }
}

// ---------------------------------------------------------------------------
// Host launcher
// ---------------------------------------------------------------------------
extern "C" void kernel_launch(void* const* d_in, const int* in_sizes, int n_in,
                              void* d_out, int out_size, void* d_ws, size_t ws_size,
                              hipStream_t stream) {
    const float* x          = (const float*)d_in[0];
    const float* ln_w       = (const float*)d_in[1];
    const float* ln_b       = (const float*)d_in[2];
    const float* in_proj_w  = (const float*)d_in[3];
    const float* conv_w     = (const float*)d_in[4];
    const float* conv_b     = (const float*)d_in[5];
    const float* x_proj_w   = (const float*)d_in[6];
    const float* dt_proj_w  = (const float*)d_in[7];
    const float* dt_proj_b  = (const float*)d_in[8];
    const float* A_log      = (const float*)d_in[9];
    const float* Dvec       = (const float*)d_in[10];
    const float* out_proj_w = (const float*)d_in[11];
    float* out = (float*)d_out;

    const int B = 4, SEQ = 1024, DIM = 256, DI = 512;
    const int ROWS = B * SEQ;                     // 4096
    const int NLD  = 112;                         // dt_rank + 2*d_state

    char* ws = (char*)d_ws;
    size_t off = 0;
    auto alloc = [&](size_t bytes) {
        size_t o = off;
        off = (off + bytes + 255) & ~(size_t)255;
        return o;
    };
    unsigned short* xn_bf  = (unsigned short*)(ws + alloc((size_t)ROWS * DIM * 2));
    unsigned short* win_bf = (unsigned short*)(ws + alloc((size_t)1024 * 256 * 2));
    unsigned short* wxp_bf = (unsigned short*)(ws + alloc((size_t)NLD * DI * 2));
    unsigned short* wop_bf = (unsigned short*)(ws + alloc((size_t)DIM * DI * 2));
    float*          xz     = (float*)(ws + alloc((size_t)ROWS * 1024 * 4));
    float*          u_f32  = (float*)(ws + alloc((size_t)ROWS * DI * 4));
    unsigned short* u_bf   = (unsigned short*)(ws + alloc((size_t)ROWS * DI * 2));
    float*          x_dbl  = (float*)(ws + alloc((size_t)ROWS * NLD * 4));
    float*          dt     = (float*)(ws + alloc((size_t)ROWS * DI * 4));
    unsigned short* y_bf   = (unsigned short*)(ws + alloc((size_t)ROWS * DI * 2));
    (void)ws_size; (void)in_sizes; (void)n_in; (void)out_size;

    // 1. Weight conversions to bf16
    {
        int n1 = 1024 * 256, n2 = NLD * DI, n3 = DIM * DI;
        f32_to_bf16<<<(n1 + 255) / 256, 256, 0, stream>>>(in_proj_w,  win_bf, n1);
        f32_to_bf16<<<(n2 + 255) / 256, 256, 0, stream>>>(x_proj_w,   wxp_bf, n2);
        f32_to_bf16<<<(n3 + 255) / 256, 256, 0, stream>>>(out_proj_w, wop_bf, n3);
    }

    // 2. LayerNorm -> bf16 (one wave per row)
    layernorm_bf16<<<(ROWS * 32 + 255) / 256, 256, 0, stream>>>(
        x, ln_w, ln_b, xn_bf, ROWS, DIM);

    // 3. in_proj: (4096x256) x (1024x256)^T -> xz (4096x1024)
    //    [WMMA bf16, LDS-staged async double-buffered B]
    gemm_bf16_lds<<<(ROWS / 128) * (1024 / 64), 256, 0, stream>>>(
        xn_bf, win_bf, xz, ROWS, 1024, 256);

    // 4. Causal depthwise conv + SiLU -> u (f32 + bf16)
    {
        int n = ROWS * DI;
        conv_silu<<<(n + 255) / 256, 256, 0, stream>>>(
            xz, conv_w, conv_b, u_f32, u_bf, B, SEQ);
    }

    // 5. x_proj: (4096x512) x (112x512)^T -> x_dbl (4096x112)   [WMMA bf16]
    {
        int tiles = (ROWS / 16) * (NLD / 16);
        gemm_bf16_nt<1><<<(tiles + 7) / 8, 256, 0, stream>>>(
            u_bf, wxp_bf, x_dbl, ROWS, NLD, DI);
    }

    // 6. dt_proj + softplus (K=16, VALU)
    {
        int n = ROWS * DI;
        dt_proj<<<(n + 255) / 256, 256, 0, stream>>>(
            x_dbl, dt_proj_w, dt_proj_b, dt, ROWS);
    }

    // 7. Selective scan (serial over t), fused D-skip + SiLU(z) gate -> y bf16
    scan_kernel<<<B * 32, 256, 0, stream>>>(
        dt, x_dbl, u_f32, A_log, Dvec, xz, y_bf, B, SEQ);

    // 8. out_proj: (4096x512) x (256x512)^T -> out (4096x256)
    //    [WMMA bf16, LDS-staged async double-buffered B]
    gemm_bf16_lds<<<(ROWS / 128) * (DIM / 64), 256, 0, stream>>>(
        y_bf, wop_bf, out, ROWS, DIM, DI);
}
